// MultiHeadedAttentionBlur_53506702573975
// MI455X (gfx1250) — matrix operand order repaired and checked by reading, not compiled, not run on hardware
//
#include <hip/hip_runtime.h>
#include <hip/hip_bf16.h>

typedef __attribute__((ext_vector_type(16))) __bf16 bf16x16;
typedef __attribute__((ext_vector_type(8)))  __bf16 bf16x8;
typedef __attribute__((ext_vector_type(4)))  __bf16 bf16x4;
typedef __attribute__((ext_vector_type(8)))  float  floatx8;
typedef __attribute__((ext_vector_type(4)))  float  floatx4;

#define NHEADS 96
#define SEQ    1024
#define HDIM   64
#define EMB    768
#define NKQ    1536   // columns of qkv we actually need: [768,2304) = k then q
#define MROWS  8192   // B*S

static __device__ __forceinline__ int refl(int i) {
  return (i < 0) ? -i : ((i > 31) ? 62 - i : i);
}

static __device__ __forceinline__ bf16x16 cat8(bf16x8 a, bf16x8 b) {
  return __builtin_shufflevector(a, b, 0,1,2,3,4,5,6,7,8,9,10,11,12,13,14,15);
}

static __device__ __forceinline__ floatx8 wmma_bf16(bf16x16 a, bf16x16 b, floatx8 c) {
  // D = A*B + C, fp32 accumulate: v_wmma_f32_16x16x32_bf16
  return __builtin_amdgcn_wmma_f32_16x16x32_bf16(false, a, false, b, (short)0, c, false, false);
}

// ---------------------------------------------------------------------------
// Kernel A: qkv projection (k,q halves only) with bf16 hi/lo error-compensated
// WMMA. C[m,f] = sum_e X[m,e] * W[768+f, e]; results split to bf16 hi/lo and
// scattered into head-major [96][1024][64] K/Q arrays.
// Block tile: 128(M) x 64(N), 8 waves as 4(M) x 2(N), each wave 32x32.
// ---------------------------------------------------------------------------
__global__ __launch_bounds__(256)
void qkv_split_kernel(const float* __restrict__ X, const float* __restrict__ W,
                      __bf16* __restrict__ Qhi, __bf16* __restrict__ Qlo,
                      __bf16* __restrict__ Khi, __bf16* __restrict__ Klo) {
  __shared__ __align__(32) __bf16 sXhi[128 * 32];
  __shared__ __align__(32) __bf16 sXlo[128 * 32];
  __shared__ __align__(32) __bf16 sWhi[64 * 32];
  __shared__ __align__(32) __bf16 sWlo[64 * 32];

  const int t  = threadIdx.x;
  const int m0 = blockIdx.y * 128;
  const int n0 = blockIdx.x * 64;

  const int w     = t >> 5;
  const int lane  = t & 31;
  const int waveM = w >> 1;        // 0..3 -> 32 rows each
  const int waveN = w & 1;         // 0..1 -> 32 cols each
  const int lrow  = lane & 15;
  const int c0a   = (lane < 16) ? 0 : 8;    // unused; A also uses c0b
  const int c0b   = (lane < 16) ? 0 : 16;   // B operand K-half select

  floatx8 acc[2][2] = {};

  for (int kk = 0; kk < EMB; kk += 32) {
    // ---- stage X tile [128 x 32] as bf16 hi/lo ----
    {
      const int r = t >> 1, half = t & 1;
      const float* src = X + (size_t)(m0 + r) * EMB + kk + half * 16;
      __bf16* dh = sXhi + r * 32 + half * 16;
      __bf16* dl = sXlo + r * 32 + half * 16;
#pragma unroll
      for (int c = 0; c < 4; ++c) {
        floatx4 v = *(const floatx4*)(src + 4 * c);
        bf16x4 h, l;
#pragma unroll
        for (int j = 0; j < 4; ++j) {
          __bf16 hb = (__bf16)v[j];
          h[j] = hb;
          l[j] = (__bf16)(v[j] - (float)hb);
        }
        *(bf16x4*)(dh + 4 * c) = h;
        *(bf16x4*)(dl + 4 * c) = l;
      }
    }
    // ---- stage W tile [64 x 32] as bf16 hi/lo (rows 768+n0 .. +64) ----
    if (t < 128) {
      const int r = t >> 1, half = t & 1;
      const float* src = W + (size_t)(EMB + n0 + r) * EMB + kk + half * 16;
      __bf16* dh = sWhi + r * 32 + half * 16;
      __bf16* dl = sWlo + r * 32 + half * 16;
#pragma unroll
      for (int c = 0; c < 4; ++c) {
        floatx4 v = *(const floatx4*)(src + 4 * c);
        bf16x4 h, l;
#pragma unroll
        for (int j = 0; j < 4; ++j) {
          __bf16 hb = (__bf16)v[j];
          h[j] = hb;
          l[j] = (__bf16)(v[j] - (float)hb);
        }
        *(bf16x4*)(dh + 4 * c) = h;
        *(bf16x4*)(dl + 4 * c) = l;
      }
    }
    __syncthreads();

    // ---- operands: contiguous k = 16*(lane/16) + s, same for A and B ----
    bf16x16 Ah[2], Al[2], Bh[2], Bl[2];
#pragma unroll
    for (int mi = 0; mi < 2; ++mi) {
      const int ar = waveM * 32 + mi * 16 + lrow;
      Ah[mi] = *(const bf16x16*)(sXhi + ar * 32 + c0b);
      Al[mi] = *(const bf16x16*)(sXlo + ar * 32 + c0b);
    }
#pragma unroll
    for (int nj = 0; nj < 2; ++nj) {
      const int bc = waveN * 32 + nj * 16 + lrow;
      Bh[nj] = *(const bf16x16*)(sWhi + bc * 32 + c0b);
      Bl[nj] = *(const bf16x16*)(sWlo + bc * 32 + c0b);
    }

#pragma unroll
    for (int mi = 0; mi < 2; ++mi)
#pragma unroll
      for (int nj = 0; nj < 2; ++nj) {
        acc[mi][nj] = wmma_bf16(Ah[mi], Bh[nj], acc[mi][nj]);  // hi*hi
        acc[mi][nj] = wmma_bf16(Ah[mi], Bl[nj], acc[mi][nj]);  // hi*lo
        acc[mi][nj] = wmma_bf16(Al[mi], Bh[nj], acc[mi][nj]);  // lo*hi
      }
    __syncthreads();
  }

  // ---- epilogue: split fp32 result to bf16 hi/lo, scatter to head-major ----
#pragma unroll
  for (int mi = 0; mi < 2; ++mi) {
#pragma unroll
    for (int nj = 0; nj < 2; ++nj) {
      const int fr = n0 + waveN * 32 + nj * 16 + lrow;  // 0..1535
      __bf16 *dh, *dl;
      int c;
      if (fr < 768) { c = fr;       dh = Khi; dl = Klo; }  // keys chunk
      else          { c = fr - 768; dh = Qhi; dl = Qlo; }  // queries chunk
      const int h = c >> 6, d = c & 63;
#pragma unroll
      for (int i = 0; i < 8; ++i) {
        const int m = m0 + waveM * 32 + mi * 16 + i + ((lane < 16) ? 0 : 8);
        const int b = m >> 10, s = m & 1023;
        const size_t idx = (((size_t)(b * 12 + h) * SEQ + s) * HDIM) + d;
        const float v = acc[mi][nj][i];
        const __bf16 hb = (__bf16)v;
        dh[idx] = hb;
        dl[idx] = (__bf16)(v - (float)hb);
      }
    }
  }
}

// ---------------------------------------------------------------------------
// Kernel B: per (head, 16-query tile): logits = SCALE * Q K^T via bf16x3 WMMA,
// 5x5 reflect gaussian blur over the 32x32 key grid, softmax over 1024 keys.
// 8 waves; wave w covers keys [w*128, w*128+128).
// ---------------------------------------------------------------------------
__global__ __launch_bounds__(256)
void attn_blur_softmax_kernel(const __bf16* __restrict__ Qhi, const __bf16* __restrict__ Qlo,
                              const __bf16* __restrict__ Khi, const __bf16* __restrict__ Klo,
                              float* __restrict__ out) {
  __shared__ __align__(16) float  sLog[16 * SEQ];  // 64 KB
  __shared__ __align__(16) float  sTmp[SEQ];       // 4 KB (per-q blur temp)
  __shared__ __align__(32) __bf16 sQh[16 * HDIM];  // 2 KB
  __shared__ __align__(32) __bf16 sQl[16 * HDIM];  // 2 KB

  const int t    = threadIdx.x;
  const int qt   = blockIdx.x;    // 0..63
  const int head = blockIdx.y;    // 0..95
  const int q0   = qt * 16;

  // stage Q tile (16 x 64, contiguous in head-major layout)
  {
    const size_t base = ((size_t)head * SEQ + q0) * HDIM;
    *(bf16x4*)(sQh + t * 4) = *(const bf16x4*)(Qhi + base + t * 4);
    *(bf16x4*)(sQl + t * 4) = *(const bf16x4*)(Qlo + base + t * 4);
  }
  __syncthreads();

  const int w    = t >> 5;
  const int lane = t & 31;
  const int lrow = lane & 15;
  const int c0a  = (lane < 16) ? 0 : 8;
  const int c0b  = (lane < 16) ? 0 : 16;

  // A operands (Q tile), two d-steps of 32
  bf16x16 Ah[2], Al[2];
#pragma unroll
  for (int s = 0; s < 2; ++s) {
    const __bf16* ph = sQh + lrow * HDIM + s * 32;
    const __bf16* pl = sQl + lrow * HDIM + s * 32;
    Ah[s] = *(const bf16x16*)(ph + c0b);
    Al[s] = *(const bf16x16*)(pl + c0b);
  }

  const __bf16* Kh = Khi + (size_t)head * SEQ * HDIM;
  const __bf16* Kl = Klo + (size_t)head * SEQ * HDIM;

  for (int kt = 0; kt < 8; ++kt) {
    const int kb = w * 128 + kt * 16;
    const __bf16* bh = Kh + (size_t)(kb + lrow) * HDIM;
    const __bf16* bl = Kl + (size_t)(kb + lrow) * HDIM;
    if (kt < 7) {  // prefetch next key tile (global_prefetch_b8)
      __builtin_prefetch(bh + 16 * HDIM, 0, 3);
      __builtin_prefetch(bl + 16 * HDIM, 0, 3);
    }
    const bf16x16 Bh0 = *(const bf16x16*)(bh + c0b);
    const bf16x16 Bh1 = *(const bf16x16*)(bh + 32 + c0b);
    const bf16x16 Bl0 = *(const bf16x16*)(bl + c0b);
    const bf16x16 Bl1 = *(const bf16x16*)(bl + 32 + c0b);

    floatx8 acc = {};
    acc = wmma_bf16(Ah[0], Bh0, acc);
    acc = wmma_bf16(Ah[1], Bh1, acc);
    acc = wmma_bf16(Ah[0], Bl0, acc);
    acc = wmma_bf16(Ah[1], Bl1, acc);
    acc = wmma_bf16(Al[0], Bh0, acc);
    acc = wmma_bf16(Al[1], Bh1, acc);

#pragma unroll
    for (int i = 0; i < 8; ++i) {
      const int q = i + ((lane < 16) ? 0 : 8);
      sLog[q * SEQ + kb + lrow] = 0.125f * acc[i];
    }
  }
  __syncthreads();

  // ---- separable 5-tap gaussian blur, reflect padding, per query row ----
  const float g0 = 0.40261996f, g1 = 0.24420135f, g2 = 0.05448868f;
  for (int q = 0; q < 16; ++q) {
    float* row = sLog + q * SEQ;
#pragma unroll
    for (int j = 0; j < 4; ++j) {  // horizontal
      const int e = t + j * 256;
      const int r = e >> 5, c = e & 31;
      const float* rr = row + r * 32;
      float s = g0 * rr[c];
      s += g1 * (rr[refl(c - 1)] + rr[refl(c + 1)]);
      s += g2 * (rr[refl(c - 2)] + rr[refl(c + 2)]);
      sTmp[e] = s;
    }
    __syncthreads();
#pragma unroll
    for (int j = 0; j < 4; ++j) {  // vertical
      const int e = t + j * 256;
      const int r = e >> 5, c = e & 31;
      float s = g0 * sTmp[r * 32 + c];
      s += g1 * (sTmp[refl(r - 1) * 32 + c] + sTmp[refl(r + 1) * 32 + c]);
      s += g2 * (sTmp[refl(r - 2) * 32 + c] + sTmp[refl(r + 2) * 32 + c]);
      row[e] = s;
    }
    __syncthreads();
  }

  // ---- softmax over 1024 keys: one wave per 2 query rows ----
#pragma unroll
  for (int qi = 0; qi < 2; ++qi) {
    const int q = 2 * w + qi;
    float v[32];
    float mx = -1e30f;
#pragma unroll
    for (int j = 0; j < 32; ++j) {
      v[j] = sLog[q * SEQ + j * 32 + lane];
      mx = fmaxf(mx, v[j]);
    }
#pragma unroll
    for (int off = 16; off > 0; off >>= 1) mx = fmaxf(mx, __shfl_xor(mx, off, 32));
    float sum = 0.f;
#pragma unroll
    for (int j = 0; j < 32; ++j) { v[j] = __expf(v[j] - mx); sum += v[j]; }
#pragma unroll
    for (int off = 16; off > 0; off >>= 1) sum += __shfl_xor(sum, off, 32);
    const float inv = 1.0f / sum;
    float* o = out + ((size_t)head * SEQ + q0 + q) * SEQ;
#pragma unroll
    for (int j = 0; j < 32; ++j) o[j * 32 + lane] = v[j] * inv;
  }
}

// ---------------------------------------------------------------------------
extern "C" void kernel_launch(void* const* d_in, const int* in_sizes, int n_in,
                              void* d_out, int out_size, void* d_ws, size_t ws_size,
                              hipStream_t stream) {
  const float* X = (const float*)d_in[0];   // [8,1024,768]
  const float* W = (const float*)d_in[1];   // [2304,768]
  float* out = (float*)d_out;               // [96,1024,1024]

  char* ws = (char*)d_ws;
  const size_t HS = (size_t)NHEADS * SEQ * HDIM;  // elements per bf16 array
  __bf16* Qhi = (__bf16*)(ws + 0 * HS);
  __bf16* Qlo = (__bf16*)(ws + 2 * HS);
  __bf16* Khi = (__bf16*)(ws + 4 * HS);
  __bf16* Klo = (__bf16*)(ws + 6 * HS);  // total 8*HS bytes = ~50.3 MB

  qkv_split_kernel<<<dim3(NKQ / 64, MROWS / 128), 256, 0, stream>>>(X, W, Qhi, Qlo, Khi, Klo);
  attn_blur_softmax_kernel<<<dim3(SEQ / 16, NHEADS), 256, 0, stream>>>(Qhi, Qlo, Khi, Klo, out);
}
